// GraphMixer_43774306681152
// MI455X (gfx1250) — matrix operand order repaired and checked by reading, not compiled, not run on hardware
//
#include <hip/hip_runtime.h>
#include <hip/hip_bf16.h>

typedef __attribute__((ext_vector_type(16))) __bf16 v16bf;
typedef __attribute__((ext_vector_type(8)))  __bf16 v8bf;
typedef __attribute__((ext_vector_type(8)))  float  v8f;

#define N_AGENTS   32
#define STATE_DIM  512
#define EMBED      64
#define NTOT       32768           // BS*T
#define SPB        16              // samples per block
#define ADJ_COLS   1024            // N_AGENTS^2
#define TOT_COLS   1088            // adj cols + 64 bias-hidden cols
#define TOT_TILES  68              // TOT_COLS / 16
#define KSTEPS_ADJ 16              // STATE_DIM / 32
#define PA_ELEMS   (TOT_TILES * KSTEPS_ADJ * 32 * 16)  // 557056 bf16
#define PW2_ELEMS  (4 * 2 * 32 * 16)                   // 4096 bf16

// ---- bf16 helpers (explicit RNE, no reliance on native cvt codegen) ----
__device__ __forceinline__ __bf16 f2bf(float f) {
    unsigned int u = __float_as_uint(f);
    unsigned int r = (u + 0x7FFFu + ((u >> 16) & 1u)) >> 16;
    union { unsigned short s; __bf16 b; } cv;
    cv.s = (unsigned short)r;
    return cv.b;
}
__device__ __forceinline__ float bf2f(__bf16 b) {
    union { __bf16 b; unsigned short s; } cv;
    cv.b = b;
    return __uint_as_float(((unsigned int)cv.s) << 16);
}

// A-fragment gather per CDNA5 16-bit A layout (16x32):
//   lane<16 : K = 0..7 (v0..3) and 16..23 (v4..7)
//   lane>=16: K = 8..15        and 24..31
// caller passes p = &row[kbase + (lane<16?0:8)], 16B-aligned.
__device__ __forceinline__ v16bf load_afrag(const __bf16* p) {
    v8bf lo = *(const v8bf*)p;
    v8bf hi = *(const v8bf*)(p + 16);
    return __builtin_shufflevector(lo, hi, 0,1,2,3,4,5,6,7,8,9,10,11,12,13,14,15);
}

// ---------------------------------------------------------------------------
// Kernel 1: repack Wadj^T || Wb1^T (512 x 1088) and W2^T (64 x 64) into bf16
// B-panels matching the WMMA B-operand layout: per (tile, kstep), lane l holds
// K = kstep*32 + (l<16?0:8) + (j<8?j:j+8)  for column N = tile*16 + (l&15).
// Panel element index: ((tile*KSTEPS + kstep)*32 + l)*16 + j  -> contiguous
// 32-byte fragment per lane.
// ---------------------------------------------------------------------------
__global__ __launch_bounds__(256) void gm_pack_kernel(
    const float* __restrict__ Wadj,   // [1024][512]
    const float* __restrict__ Wb1,    // [64][512]
    const float* __restrict__ W2,     // [64][64]
    __bf16* __restrict__ panelAdj,    // [PA_ELEMS]
    __bf16* __restrict__ panelW2)     // [PW2_ELEMS]
{
    int idx = blockIdx.x * 256 + threadIdx.x;
    if (idx < PA_ELEMS) {
        int j  = idx & 15;
        int l  = (idx >> 4) & 31;
        int ks = (idx >> 9) & 15;
        int t  = idx >> 13;
        int k  = ks * 32 + ((l < 16) ? 0 : 8) + ((j < 8) ? j : j + 8);
        int col = t * 16 + (l & 15);
        float v = (col < ADJ_COLS) ? Wadj[col * STATE_DIM + k]
                                   : Wb1[(col - ADJ_COLS) * STATE_DIM + k];
        panelAdj[idx] = f2bf(v);
    } else if (idx < PA_ELEMS + PW2_ELEMS) {
        int q  = idx - PA_ELEMS;
        int j  = q & 15;
        int l  = (q >> 4) & 31;
        int ks = (q >> 9) & 1;
        int tn = q >> 10;
        int k  = ks * 32 + ((l < 16) ? 0 : 8) + ((j < 8) ? j : j + 8);
        int col = tn * 16 + (l & 15);
        panelW2[q] = f2bf(W2[col * EMBED + k]);  // B[e][f] = W2[f][e]
    }
}

// ---------------------------------------------------------------------------
// Kernel 2: fully fused per-16-sample pipeline.
//   Phase B: x1 = relu(qs*W1+b1) -> WMMA GEMM with W2^T -> relu -> r[a]=x2.Wr
//   Phase C: WMMA GEMM states x (Wadj^T||Wb1^T) -> sigmoid adj (diag masked,
//            degree accum) + relu bias-hidden dotted with Wb2
//   Phase D: c[j] = sum_i adj[i][j]*inv_deg[i];
//            out = sum_j (1+c[j])*r[j] + 32*br + bias + bb2
// ---------------------------------------------------------------------------
__global__ __launch_bounds__(256) void gm_fused_kernel(
    const float* __restrict__ qs_g,     // [NTOT][32]
    const float* __restrict__ st_g,     // [NTOT][512]
    const float* __restrict__ W1_g,     // [64]
    const float* __restrict__ b1_g,     // [64]
    const float* __restrict__ b2_g,     // [64]
    const float* __restrict__ badj_g,   // [1024]
    const float* __restrict__ Wr_g,     // [64]
    const float* __restrict__ br_g,     // [1]
    const float* __restrict__ bb1_g,    // [64]
    const float* __restrict__ Wb2_g,    // [64]
    const float* __restrict__ bb2_g,    // [1]
    const __bf16* __restrict__ panelAdj,
    const __bf16* __restrict__ panelW2,
    float* __restrict__ out_g)          // [NTOT]
{
    __shared__ float  qsL[SPB * N_AGENTS];        // 512 f32, agents flattened
    __shared__ __bf16 stB[SPB * STATE_DIM];       // 16 KB, A-tile (states)
    __shared__ __bf16 buf[SPB * ADJ_COLS];        // 32 KB: x1-half / adj alias
    __shared__ float  rAcc[SPB * N_AGENTS];       // r[j] per sample
    __shared__ float  degL[SPB * N_AGENTS];       // deg[i] -> later 1/(deg+eps)
    __shared__ float  biasL[SPB];
    __shared__ float  outL[SPB];

    const int tid  = threadIdx.x;
    const int lane = tid & 31;
    // wave-uniform scalar: lets the compiler emit scalar loops (no EXEC
    // masking around WMMA clauses)
    const int wave = __builtin_amdgcn_readfirstlane(tid >> 5);
    const int n0   = blockIdx.x * SPB;
    const int lhalf = (lane < 16) ? 0 : 8;   // A-frag K offset / D-row offset
    const int ln15  = lane & 15;

    // ---- init + async qs -> LDS (overlapped with states bf16 convert) ----
    {
        const float* qsrc = qs_g + (size_t)n0 * N_AGENTS;
#pragma unroll
        for (int i = tid; i < SPB * N_AGENTS; i += 256) {
            unsigned ldsOff = (unsigned)(uintptr_t)&qsL[i];  // low 32b = LDS addr
            unsigned voff   = (unsigned)(i * 4);
            asm volatile("global_load_async_to_lds_b32 %0, %1, %2"
                         :: "v"(ldsOff), "v"(voff), "s"(qsrc) : "memory");
        }
    }
    for (int i = tid; i < SPB * N_AGENTS; i += 256) { rAcc[i] = 0.f; degL[i] = 0.f; }
    if (tid < SPB) { biasL[tid] = 0.f; outL[tid] = 0.f; }
    for (int i = tid; i < SPB * STATE_DIM; i += 256)
        stB[i] = f2bf(st_g[(size_t)n0 * STATE_DIM + i]);
    asm volatile("s_wait_asynccnt 0x0" ::: "memory");
    __syncthreads();

    // ---- Phase B: embed MLP, two halves of 256 rows (8 samples each) ----
    for (int h = 0; h < 2; ++h) {
        // x1 = relu(qs*W1 + b1) into buf[256][64] bf16
        for (int i = tid; i < 256 * EMBED; i += 256) {
            int row = i >> 6;             // local row 0..255
            int e   = i & 63;
            float v = qsL[h * 256 + row] * W1_g[e] + b1_g[e];
            buf[i] = f2bf(fmaxf(v, 0.f));
        }
        __syncthreads();

        // GEMM [256 x 64] @ [64 x 64]: 16 M-tiles x 4 N-tiles over 8 waves
        for (int tile = wave; tile < 64; tile += 8) {
            int mt = tile >> 2, nt = tile & 3;
            v8f acc = {0.f, 0.f, 0.f, 0.f, 0.f, 0.f, 0.f, 0.f};
#pragma unroll
            for (int ks = 0; ks < 2; ++ks) {
                int rowA = mt * 16 + ln15;
                v16bf a = load_afrag(&buf[rowA * EMBED + ks * 32 + lhalf]);
                v16bf b = *(const v16bf*)(panelW2 + (((nt * 2 + ks) * 32 + lane) << 4));
                acc = __builtin_amdgcn_wmma_f32_16x16x32_bf16(
                        false, a, false, b, (short)0, acc, false, false);
            }
            int   col = nt * 16 + ln15;
            float wr  = Wr_g[col];
            float bb  = b2_g[col];
            int rbase = h * 256 + mt * 16 + lhalf;
#pragma unroll
            for (int m = 0; m < 8; ++m) {
                float x2 = fmaxf(acc[m] + bb, 0.f);
                atomicAdd(&rAcc[rbase + m], x2 * wr);
            }
        }
        __syncthreads();   // buf fully consumed before refill / adj reuse
    }

    // ---- Phase C: adjacency + bias-head GEMM [16 x 512] @ [512 x 1088] ----
    for (int t = wave; t < TOT_TILES; t += 8) {
        v8f acc = {0.f, 0.f, 0.f, 0.f, 0.f, 0.f, 0.f, 0.f};
#pragma unroll
        for (int ks = 0; ks < KSTEPS_ADJ; ++ks) {
            v16bf a = load_afrag(&stB[ln15 * STATE_DIM + ks * 32 + lhalf]);
            v16bf b = *(const v16bf*)(panelAdj + (((t * KSTEPS_ADJ + ks) * 32 + lane) << 4));
            acc = __builtin_amdgcn_wmma_f32_16x16x32_bf16(
                    false, a, false, b, (short)0, acc, false, false);
        }
        int col = t * 16 + ln15;
        if (t < 64) {
            float bj = badj_g[col];
            int ii = col >> 5, jj = col & 31;
#pragma unroll
            for (int m = 0; m < 8; ++m) {
                int s = lhalf + m;                       // sample row 0..15
                float z  = acc[m] + bj;
                float sg = 1.f / (1.f + __expf(-z));
                if (ii == jj) sg = 0.f;                  // mask self-loops
                buf[s * ADJ_COLS + col] = f2bf(sg);
                atomicAdd(&degL[s * N_AGENTS + ii], sg);
            }
        } else {
            int e = col - ADJ_COLS;
            float be = bb1_g[e];
            float we = Wb2_g[e];
#pragma unroll
            for (int m = 0; m < 8; ++m) {
                int s = lhalf + m;
                float hv = fmaxf(acc[m] + be, 0.f);
                atomicAdd(&biasL[s], hv * we);
            }
        }
    }
    __syncthreads();

    // ---- reciprocal degrees (one divide per (s,i) instead of 32 per pair) --
    for (int i = tid; i < SPB * N_AGENTS; i += 256)
        degL[i] = 1.f / (degL[i] + 1e-6f);
    __syncthreads();

    // ---- Phase D: c[j] and readout reduction ----
    for (int p = tid; p < SPB * N_AGENTS; p += 256) {
        int s = p >> 5, j = p & 31;
        float cj = 0.f;
#pragma unroll 4
        for (int i = 0; i < N_AGENTS; ++i) {
            float av = bf2f(buf[s * ADJ_COLS + i * N_AGENTS + j]);
            cj += av * degL[s * N_AGENTS + i];
        }
        atomicAdd(&outL[s], (1.f + cj) * rAcc[p]);
    }
    __syncthreads();

    if (tid < SPB) {
        out_g[n0 + tid] = outL[tid] + biasL[tid]
                        + (float)N_AGENTS * br_g[0] + bb2_g[0];
    }
}

extern "C" void kernel_launch(void* const* d_in, const int* in_sizes, int n_in,
                              void* d_out, int out_size, void* d_ws, size_t ws_size,
                              hipStream_t stream) {
    const float* qs   = (const float*)d_in[0];
    const float* st   = (const float*)d_in[1];
    const float* W1   = (const float*)d_in[2];
    const float* b1   = (const float*)d_in[3];
    const float* W2   = (const float*)d_in[4];
    const float* b2   = (const float*)d_in[5];
    const float* Wadj = (const float*)d_in[6];
    const float* badj = (const float*)d_in[7];
    const float* Wr   = (const float*)d_in[8];
    const float* br   = (const float*)d_in[9];
    const float* Wb1  = (const float*)d_in[10];
    const float* bb1  = (const float*)d_in[11];
    const float* Wb2  = (const float*)d_in[12];
    const float* bb2  = (const float*)d_in[13];
    float* out = (float*)d_out;

    __bf16* panelAdj = (__bf16*)d_ws;
    __bf16* panelW2  = panelAdj + PA_ELEMS;

    int packTotal  = PA_ELEMS + PW2_ELEMS;
    int packBlocks = (packTotal + 255) / 256;
    gm_pack_kernel<<<packBlocks, 256, 0, stream>>>(Wadj, Wb1, W2, panelAdj, panelW2);

    int blocks = NTOT / SPB;   // 2048
    gm_fused_kernel<<<blocks, 256, 0, stream>>>(
        qs, st, W1, b1, b2, badj, Wr, br, bb1, Wb2, bb2,
        panelAdj, panelW2, out);
}
